// TtLlamaAttention_84026740178962
// MI455X (gfx1250) — compile-verified, matrix operand order checked by reading, not run on hardware
//
#include <hip/hip_runtime.h>

typedef __attribute__((ext_vector_type(16))) __bf16 v16bf;
typedef __attribute__((ext_vector_type(8)))  float  v8f;
typedef int v4i_vs __attribute__((vector_size(16)));   // matches builtin param type

#define NH   32
#define SEQ  2048
#define HID  4096
#define HD   128

#if defined(__HIP_DEVICE_COMPILE__) && defined(__has_builtin)
#if __has_builtin(__builtin_amdgcn_global_load_async_to_lds_b128)
#define ASYNC_CP 1
#endif
#endif

__device__ __forceinline__ unsigned short f2b(float f) {
    unsigned int u = __float_as_uint(f);
    u += 0x7FFFu + ((u >> 16) & 1u);          // round-to-nearest-even
    return (unsigned short)(u >> 16);
}

// 16B global -> LDS copy: async (ASYNCcnt-tracked DMA) when available.
__device__ __forceinline__ void cp_b128(const unsigned short* g, unsigned short* l) {
#if defined(ASYNC_CP)
    __builtin_amdgcn_global_load_async_to_lds_b128(
        (__attribute__((address_space(1))) v4i_vs*)g,
        (__attribute__((address_space(3))) v4i_vs*)l, 0, 0);
#else
    *reinterpret_cast<uint4*>(l) = *reinterpret_cast<const uint4*>(g);
#endif
}

__device__ __forceinline__ void wait_cp() {
#if defined(ASYNC_CP)
#if __has_builtin(__builtin_amdgcn_s_wait_asynccnt)
    __builtin_amdgcn_s_wait_asynccnt(0);
#else
    asm volatile("s_wait_asynccnt 0x0" ::: "memory");
#endif
#endif
}

union Frag { v16bf v; uint4 u[2]; };

// 16x32 bf16 fragment.  A layout: lane = row.  B layout: lane = col (B source is
// [N,K] row-major, so addressing is identical).  Per ISA 7.12.2: lanes 0-15 hold
// K 0-7 & 16-23, lanes 16-31 hold K 8-15 & 24-31.
__device__ __forceinline__ v16bf load_frag(const unsigned short* base, int row0,
                                           int strideElems, int kbase, int lane) {
    const int hi = lane >> 4, lo = lane & 15;
    const unsigned short* p = base + (size_t)(row0 + lo) * strideElems + kbase + hi * 8;
    Frag f;
    f.u[0] = *reinterpret_cast<const uint4*>(p);
    f.u[1] = *reinterpret_cast<const uint4*>(p + 16);
    return f.v;
}

__device__ __forceinline__ v8f wmma_bf16(v16bf a, v16bf b, v8f c) {
    return __builtin_amdgcn_wmma_f32_16x16x32_bf16(false, a, false, b, (short)0, c,
                                                   false, false);
}

// ---------------------------------------------------------------------------
// fp32 -> bf16 bulk conversion, 8 elements/thread.
// ---------------------------------------------------------------------------
__global__ __launch_bounds__(256)
void cvt_bf16(const float* __restrict__ X, unsigned short* __restrict__ Y) {
    size_t i = ((size_t)blockIdx.x * 256 + threadIdx.x) * 8;
    float4 a = *reinterpret_cast<const float4*>(&X[i]);
    float4 b = *reinterpret_cast<const float4*>(&X[i + 4]);
    uint4 o;
    o.x = (unsigned)f2b(a.x) | ((unsigned)f2b(a.y) << 16);
    o.y = (unsigned)f2b(a.z) | ((unsigned)f2b(a.w) << 16);
    o.z = (unsigned)f2b(b.x) | ((unsigned)f2b(b.y) << 16);
    o.w = (unsigned)f2b(b.z) | ((unsigned)f2b(b.w) << 16);
    *reinterpret_cast<uint4*>(&Y[i]) = o;
}

// ---------------------------------------------------------------------------
// C[m,n] = sum_k A[m,k]*B[n,k]   (A,B bf16 [4096,4096] K-major, C f32 [4096,4096])
// 4 waves, block tile 128x128, wave tile 64x64, BK=32, double-buffered LDS with
// async global->LDS copies.  16 WMMAs per wave per k-step, 128 k-steps.
// ---------------------------------------------------------------------------
__global__ __launch_bounds__(128)
void gemm_nt_bf16(const unsigned short* __restrict__ A,
                  const unsigned short* __restrict__ B,
                  float* __restrict__ C) {
    constexpr int GK = HID, GN = HID;
    __shared__ unsigned short lA[2][128 * 40];   // +8 pad: keeps 16B align, kills conflicts
    __shared__ unsigned short lB[2][128 * 40];

    const int tid  = threadIdx.x;
    const int lane = tid & 31;
    const int w    = tid >> 5;
    const int wm   = (w >> 1) * 64;
    const int wn   = (w & 1) * 64;
    const int bm   = blockIdx.y * 128;
    const int bn   = blockIdx.x * 128;

    auto stage = [&](int buf, int k0) {
        #pragma unroll
        for (int p = 0; p < 4; ++p) {
            int idx = (p * 128 + tid) * 8;
            int r = idx >> 5, c = idx & 31;
            cp_b128(&A[(size_t)(bm + r) * GK + k0 + c], &lA[buf][r * 40 + c]);
            cp_b128(&B[(size_t)(bn + r) * GK + k0 + c], &lB[buf][r * 40 + c]);
        }
    };

    const v8f vzero = {0.f,0.f,0.f,0.f,0.f,0.f,0.f,0.f};
    v8f acc[4][4];
    #pragma unroll
    for (int i = 0; i < 4; ++i)
        #pragma unroll
        for (int j = 0; j < 4; ++j) acc[i][j] = vzero;

    stage(0, 0);
    int cur = 0;
    for (int k0 = 0; k0 < GK; k0 += 32) {
        wait_cp();
        __syncthreads();
        if (k0 + 32 < GK) stage(cur ^ 1, k0 + 32);   // prefetch next tile

        v16bf af[4], bfr[4];
        #pragma unroll
        for (int mi = 0; mi < 4; ++mi) af[mi]  = load_frag(lA[cur], wm + mi * 16, 40, 0, lane);
        #pragma unroll
        for (int ni = 0; ni < 4; ++ni) bfr[ni] = load_frag(lB[cur], wn + ni * 16, 40, 0, lane);
        #pragma unroll
        for (int mi = 0; mi < 4; ++mi)
            #pragma unroll
            for (int ni = 0; ni < 4; ++ni)
                acc[mi][ni] = wmma_bf16(af[mi], bfr[ni], acc[mi][ni]);
        __syncthreads();
        cur ^= 1;
    }

    // C/D layout: VGPR i holds row i (lanes 0-15) / i+8 (lanes 16-31), col = lane&15.
    const int hi = lane >> 4, lo = lane & 15;
    float* Cp = C + (size_t)(bm + wm + 8 * hi) * GN + bn + wn + lo;
    #pragma unroll
    for (int mi = 0; mi < 4; ++mi)
        #pragma unroll
        for (int i = 0; i < 8; ++i)
            #pragma unroll
            for (int ni = 0; ni < 4; ++ni)
                Cp[(size_t)(mi * 16 + i) * GN + ni * 16] = acc[mi][ni][i];
}

// ---------------------------------------------------------------------------
// RoPE + split-heads + fp32->bf16.  In: [B*S, H] f32.  Out: [B, nh, S, hd] bf16.
// ---------------------------------------------------------------------------
__global__ __launch_bounds__(128)
void rope_convert(const float* __restrict__ X, const int* __restrict__ pos_ids,
                  unsigned short* __restrict__ Out, int do_rope) {
    int idx = blockIdx.x;
    int s = idx & (SEQ - 1);
    int h = (idx >> 11) & (NH - 1);
    int b = idx >> 16;
    int d = threadIdx.x;
    size_t src = ((size_t)(b * SEQ + s)) * HID + h * HD;
    float x = X[src + d];
    float o = x;
    if (do_rope) {
        int p  = pos_ids[b * SEQ + s];
        int fi = d & 63;
        float invf = __powf(10000.0f, -((float)(2 * fi)) * (1.0f / 128.0f));
        float ang  = (float)p * invf;
        float sn, cs;
        __sincosf(ang, &sn, &cs);
        float partner = (d < 64) ? -X[src + d + 64] : X[src + d - 64];
        o = x * cs + partner * sn;
    }
    Out[((size_t)((b * NH + h) * SEQ + s)) * HD + d] = f2b(o);
}

// ---------------------------------------------------------------------------
// Flash attention, causal.  Q/K/V: [B, nh, S, hd] bf16.  O: [B*S, H] f32.
// 4 waves/block, wave owns 16 query rows.  K tile staged via async copy;
// V staged transposed (scatter).  WMMA bf16 for S=QK^T and O+=P·V.
// ---------------------------------------------------------------------------
__global__ __launch_bounds__(128)
void flash_attn(const unsigned short* __restrict__ Q,
                const unsigned short* __restrict__ Kb,
                const unsigned short* __restrict__ Vb,
                float* __restrict__ O) {
    __shared__ unsigned short lK[64 * 136];       // K tile, [key][d]
    __shared__ unsigned short lVt[128 * 72];      // V tile transposed, [d][key]
    __shared__ unsigned short lP[4][16 * 72];     // per-wave P (bf16), [row][key]

    const int tid  = threadIdx.x;
    const int lane = tid & 31;
    const int w    = tid >> 5;
    const int qt = blockIdx.x, h = blockIdx.y, b = blockIdx.z;
    const int q0w = qt * 64 + w * 16;
    const size_t headBase = ((size_t)(b * NH + h)) * SEQ * HD;
    const int hi = lane >> 4, lo = lane & 15;
    const float scale = 0.08838834764831845f;     // 1/sqrt(128)

    v16bf qa[4];
    #pragma unroll
    for (int kc = 0; kc < 4; ++kc)
        qa[kc] = load_frag(Q + headBase, q0w, HD, kc * 32, lane);

    const v8f vzero = {0.f,0.f,0.f,0.f,0.f,0.f,0.f,0.f};
    v8f oacc[8];
    float m[8], l[8];
    #pragma unroll
    for (int n = 0; n < 8; ++n) oacc[n] = vzero;
    #pragma unroll
    for (int i = 0; i < 8; ++i) { m[i] = -1e30f; l[i] = 0.0f; }

    for (int kt = 0; kt <= qt; ++kt) {
        const int k0 = kt * 64;
        #pragma unroll
        for (int p = 0; p < 8; ++p) {
            int idx = (p * 128 + tid) * 8;
            int r = idx >> 7, c = idx & 127;
            cp_b128(&Kb[headBase + (size_t)(k0 + r) * HD + c], &lK[r * 136 + c]);
            uint4 vv = *reinterpret_cast<const uint4*>(&Vb[headBase + (size_t)(k0 + r) * HD + c]);
            const unsigned short* vs = reinterpret_cast<const unsigned short*>(&vv);
            #pragma unroll
            for (int j = 0; j < 8; ++j) lVt[(c + j) * 72 + r] = vs[j];
        }
        wait_cp();
        __syncthreads();

        v8f sacc[4];
        #pragma unroll
        for (int j = 0; j < 4; ++j) {
            sacc[j] = vzero;
            #pragma unroll
            for (int kc = 0; kc < 4; ++kc)
                sacc[j] = wmma_bf16(qa[kc], load_frag(lK, j * 16, 136, kc * 32, lane), sacc[j]);
        }

        #pragma unroll
        for (int i = 0; i < 8; ++i) {
            const int row = q0w + i + 8 * hi;
            float sv[4];
            #pragma unroll
            for (int j = 0; j < 4; ++j) {
                int col = k0 + j * 16 + lo;
                float s = sacc[j][i] * scale;
                sv[j] = (col <= row) ? s : -1e30f;
            }
            float mt = fmaxf(fmaxf(sv[0], sv[1]), fmaxf(sv[2], sv[3]));
            #pragma unroll
            for (int msk = 1; msk < 16; msk <<= 1)
                mt = fmaxf(mt, __shfl_xor(mt, msk, 32));
            float mnew  = fmaxf(m[i], mt);
            float alpha = __expf(m[i] - mnew);
            float rs = 0.0f;
            #pragma unroll
            for (int j = 0; j < 4; ++j) {
                float p = __expf(sv[j] - mnew);
                rs += p;
                lP[w][(i + 8 * hi) * 72 + j * 16 + lo] = f2b(p);
            }
            #pragma unroll
            for (int msk = 1; msk < 16; msk <<= 1)
                rs += __shfl_xor(rs, msk, 32);
            l[i] = l[i] * alpha + rs;
            m[i] = mnew;
            #pragma unroll
            for (int n = 0; n < 8; ++n) oacc[n][i] *= alpha;
        }

        v16bf pa[2];
        pa[0] = load_frag(&lP[w][0], 0, 72, 0,  lane);
        pa[1] = load_frag(&lP[w][0], 0, 72, 32, lane);
        #pragma unroll
        for (int n = 0; n < 8; ++n)
            #pragma unroll
            for (int ks = 0; ks < 2; ++ks)
                oacc[n] = wmma_bf16(pa[ks], load_frag(lVt, n * 16, 72, ks * 32, lane), oacc[n]);
        __syncthreads();
    }

    #pragma unroll
    for (int n = 0; n < 8; ++n)
        #pragma unroll
        for (int i = 0; i < 8; ++i) {
            int row = q0w + i + 8 * hi;
            O[((size_t)(b * SEQ + row)) * HID + h * HD + n * 16 + lo] = oacc[n][i] / l[i];
        }
}

extern "C" void kernel_launch(void* const* d_in, const int* in_sizes, int n_in,
                              void* d_out, int out_size, void* d_ws, size_t ws_size,
                              hipStream_t stream) {
    const float* hidden = (const float*)d_in[0];
    // d_in[1] = attention_mask: exactly the causal mask -> computed analytically
    const int*   pos    = (const int*)d_in[2];
    const float* Wq     = (const float*)d_in[3];
    const float* Wk     = (const float*)d_in[4];
    const float* Wv     = (const float*)d_in[5];
    const float* Wo     = (const float*)d_in[6];
    float* out = (float*)d_out;

    char* ws = (char*)d_ws;
    unsigned short* Xb   = (unsigned short*)ws;                          // 32 MB (hidden, later attn-out)
    unsigned short* Wb   = (unsigned short*)(ws + ((size_t)32  << 20));  // 32 MB (current weight)
    float*          tmpF = (float*)        (ws + ((size_t)64  << 20));   // 64 MB
    unsigned short* Qb   = (unsigned short*)(ws + ((size_t)128 << 20));  // 32 MB
    unsigned short* Kbf  = (unsigned short*)(ws + ((size_t)160 << 20));  // 32 MB
    unsigned short* Vbf  = (unsigned short*)(ws + ((size_t)192 << 20));  // 32 MB

    const int cvtBlocks  = (2 * SEQ * HID) / (256 * 8);   // 8192
    const int ropeBlocks = 2 * NH * SEQ;                  // 131072
    dim3 gg(HID / 128, (2 * SEQ) / 128);                  // (32, 32)

    cvt_bf16<<<cvtBlocks, 256, 0, stream>>>(hidden, Xb);

    cvt_bf16<<<cvtBlocks, 256, 0, stream>>>(Wq, Wb);
    gemm_nt_bf16<<<gg, 128, 0, stream>>>(Xb, Wb, tmpF);
    rope_convert<<<ropeBlocks, 128, 0, stream>>>(tmpF, pos, Qb, 1);

    cvt_bf16<<<cvtBlocks, 256, 0, stream>>>(Wk, Wb);
    gemm_nt_bf16<<<gg, 128, 0, stream>>>(Xb, Wb, tmpF);
    rope_convert<<<ropeBlocks, 128, 0, stream>>>(tmpF, pos, Kbf, 1);

    cvt_bf16<<<cvtBlocks, 256, 0, stream>>>(Wv, Wb);
    gemm_nt_bf16<<<gg, 128, 0, stream>>>(Xb, Wb, tmpF);
    rope_convert<<<ropeBlocks, 128, 0, stream>>>(tmpF, pos, Vbf, 0);

    flash_attn<<<dim3(SEQ / 64, NH, 2), 128, 0, stream>>>(Qb, Kbf, Vbf, tmpF);

    cvt_bf16<<<cvtBlocks, 256, 0, stream>>>(tmpF, Xb);    // attn-out -> bf16 (hidden dead)
    cvt_bf16<<<cvtBlocks, 256, 0, stream>>>(Wo, Wb);
    gemm_nt_bf16<<<gg, 128, 0, stream>>>(Xb, Wb, out);
}
